// DynamicConv2D_53652731461889
// MI455X (gfx1250) — matrix operand order repaired and checked by reading, not compiled, not run on hardware
//
#include <hip/hip_runtime.h>
#include <hip/hip_bf16.h>

typedef __attribute__((ext_vector_type(16))) __bf16 v16bf;
typedef __attribute__((ext_vector_type(8)))  float  v8f;

#define B_   32
#define H_   64
#define W_   64
#define CIN  128
#define F_   128
#define NK   4
#define NR   4

// ---------------------------------------------------------------- routing ---
__global__ void routing_kernel(const float* __restrict__ x,
                               const float* __restrict__ Wr,   // [128,4]
                               const float* __restrict__ Wa,   // [4,4]
                               float* __restrict__ attn) {     // [32,4]
    __shared__ float s_part[256];
    __shared__ float s_pool[128];
    __shared__ float s_red[NR];
    const int b = blockIdx.x, t = threadIdx.x;
    const int c = t & 127, half = t >> 7;
    const float* xb = x + (size_t)b * H_ * W_ * CIN;
    float s = 0.f;
    for (int p = half; p < H_ * W_; p += 2) s += xb[(size_t)p * CIN + c];
    s_part[t] = s;
    __syncthreads();
    if (t < 128) s_pool[t] = (s_part[t] + s_part[t + 128]) * (1.0f / (H_ * W_));
    __syncthreads();
    if (t < NR) {
        float acc = 0.f;
        for (int cc = 0; cc < CIN; ++cc) acc += s_pool[cc] * Wr[cc * NR + t];
        s_red[t] = fmaxf(acc, 0.f);
    }
    __syncthreads();
    if (t == 0) {
        float lg[NK], mx = -1e30f;
        for (int k = 0; k < NK; ++k) {
            float a = 0.f;
            for (int r = 0; r < NR; ++r) a += s_red[r] * Wa[r * NK + k];
            a *= (1.0f / 30.0f);
            lg[k] = a; mx = fmaxf(mx, a);
        }
        float den = 0.f, e[NK];
        for (int k = 0; k < NK; ++k) { e[k] = __expf(lg[k] - mx); den += e[k]; }
        float inv = 1.0f / den;
        for (int k = 0; k < NK; ++k) attn[b * NK + k] = e[k] * inv;
    }
}

// ------------------------------------------------- mix experts + transpose ---
// Wt layout: [b][tap][f][ci] bf16  (transposed so WMMA B reads contiguous K)
__global__ void mix_kernel(const float* __restrict__ Wk,   // [4,3,3,128,128]
                           const float* __restrict__ attn,
                           __hip_bfloat16* __restrict__ Wt) {
    __shared__ __hip_bfloat16 sT[128][132];     // [f][ci] padded
    const int b = blockIdx.x / 9;
    const int tap = blockIdx.x - b * 9;
    const float a0 = attn[b * 4 + 0], a1 = attn[b * 4 + 1];
    const float a2 = attn[b * 4 + 2], a3 = attn[b * 4 + 3];
    const size_t E = (size_t)9 * 128 * 128;
    const float* w0 = Wk + (size_t)tap * 128 * 128;       // expert 0, this tap
    for (int i = threadIdx.x; i < 128 * 128; i += 256) {  // i = ci*128 + f
        float v = a0 * w0[i] + a1 * w0[i + E] + a2 * w0[i + 2 * E] + a3 * w0[i + 3 * E];
        sT[i & 127][i >> 7] = __float2bfloat16(v);        // sT[f][ci]
    }
    __syncthreads();
    __hip_bfloat16* outp = Wt + ((size_t)b * 9 + tap) * (128 * 128);
    for (int i = threadIdx.x; i < 128 * 128; i += 256)    // i = f*128 + ci
        outp[i] = sT[i >> 7][i & 127];
}

// -------------------------------------------------- fold bias + BN affine ---
__global__ void prep_kernel(const float* __restrict__ attn,
                            const float* __restrict__ bias,   // [4,128]
                            const float* __restrict__ scale,
                            const float* __restrict__ bnb,
                            const float* __restrict__ mean,
                            const float* __restrict__ var,
                            float* __restrict__ beta,         // [32,128]
                            float* __restrict__ alpha) {      // [128]
    const int b = blockIdx.x, f = threadIdx.x;
    float inv = scale[f] * rsqrtf(var[f] + 1e-5f);
    float bm = 0.f;
    for (int k = 0; k < NK; ++k) bm += attn[b * 4 + k] * bias[k * 128 + f];
    beta[b * 128 + f] = (bm - mean[f]) * inv + bnb[f];
    if (b == 0) alpha[f] = inv;
}

// ------------------------------------------------------- WMMA conv kernel ---
// grid = (2 strips, 32 row-pairs, 32 samples), 256 threads = 8 waves.
// Block tile: M=64 (2 rows x 32 cols) x N=128. Wave: 2 M-tiles x 2 N-tiles
// (4 v8f accumulators), so each k-step = 4 fragment loads -> 4 WMMAs (1:1).
// Weights staged in half-K slices (128f x 64ci) to fit 64KB static LDS.
__global__ __launch_bounds__(256) void condconv_wmma_kernel(
        const float* __restrict__ x,
        const __hip_bfloat16* __restrict__ Wt,   // [B][9][F][CIN] bf16
        const float* __restrict__ alpha,         // [F]
        const float* __restrict__ beta,          // [B][F]
        float* __restrict__ out) {
    __shared__ __align__(16) __hip_bfloat16 sIn[4][34][136];  // 36,992 B
    __shared__ __align__(16) __hip_bfloat16 sW[128][72];      // 18,432 B [f][ci%64]

    const int strip = blockIdx.x;          // 0..1  (32 output cols each)
    const int hp    = blockIdx.y;          // 0..31 (output rows 2hp, 2hp+1)
    const int b     = blockIdx.z;          // 0..31
    const int h0    = hp * 2;
    const int tid   = threadIdx.x;
    const int lane  = tid & 31;
    const int wid   = tid >> 5;            // 0..7
    const int g     = wid & 1;             // M-group: output row h0+g
    const int q     = wid >> 1;            // N-group: f base q*32
    const int nb    = q * 32;
    const int l16   = lane & 15;
    const int halfl = lane >> 4;

    const float* xb = x + (size_t)b * H_ * W_ * CIN;

    // ---- stage input rows h0-1..h0+2, cols strip*32-1..+32, as bf16 --------
    // 4*34 = 136 pixels x 32 float4 chunks = 4352 = 17*256 (uniform trips)
    for (int idx = tid; idx < 136 * 32; idx += 256) {
        int pix = idx >> 5, chunk = idx & 31;
        int r = pix / 34, cpos = pix - r * 34;
        int hh = h0 - 1 + r;
        int ww = strip * 32 + cpos - 1;
        float4 v = make_float4(0.f, 0.f, 0.f, 0.f);
        if (hh >= 0 && hh < H_ && (unsigned)ww < (unsigned)W_)
            v = *reinterpret_cast<const float4*>(
                    xb + (((size_t)hh * W_ + ww) * CIN + chunk * 4));
        union { __hip_bfloat16 e[4]; uint2 u; } pk;
        pk.e[0] = __float2bfloat16(v.x);
        pk.e[1] = __float2bfloat16(v.y);
        pk.e[2] = __float2bfloat16(v.z);
        pk.e[3] = __float2bfloat16(v.w);
        *reinterpret_cast<uint2*>(&sIn[r][cpos][chunk * 4]) = pk.u;
    }

    v8f c00 = {}, c01 = {}, c10 = {}, c11 = {};
    const __hip_bfloat16* wtb = Wt + (size_t)b * 9 * (F_ * CIN);

    // 18 stages: (tap 0..8) x (K-half 0..1)
    for (int s = 0; s < 18; ++s) {
        const int tap = s >> 1;
        const int hk  = s & 1;               // ci base hk*64
        __syncthreads();                     // previous stage's sW reads done
        // stage 128 rows x 64 ci of this tap's weights: 1024 = 4*256 uint4s
        {
            const __hip_bfloat16* wsrc = wtb + (size_t)tap * (F_ * CIN) + hk * 64;
            for (int idx = tid; idx < 1024; idx += 256) {
                int f = idx >> 3, qq = idx & 7;      // 8 uint4 per f-row
                *reinterpret_cast<uint4*>(&sW[f][qq * 8]) =
                    *reinterpret_cast<const uint4*>(wsrc + (size_t)f * CIN + qq * 8);
            }
        }
        __syncthreads();

        const int kh = tap / 3, kw = tap - kh * 3;
        const int rIn = g + kh;              // LDS input row for this wave
        #pragma unroll
        for (int ks = 0; ks < 2; ++ks) {     // 2 k-steps of 32 within the half
            const int kb = ks * 32;
            union { uint4 u[2]; v16bf v; } a0, a1, b0, b1;
            // A fragments (lane=M=col, slots: K = half*8 + j (+16 for hi 8))
            const int ciA = hk * 64 + kb + halfl * 8;
            const __hip_bfloat16* ap0 = &sIn[rIn][l16 + kw][ciA];
            a0.u[0] = *reinterpret_cast<const uint4*>(ap0);
            a0.u[1] = *reinterpret_cast<const uint4*>(ap0 + 16);
            const __hip_bfloat16* ap1 = &sIn[rIn][16 + l16 + kw][ciA];
            a1.u[0] = *reinterpret_cast<const uint4*>(ap1);
            a1.u[1] = *reinterpret_cast<const uint4*>(ap1 + 16);
            // B fragments (lane=N=f, slots: K = half*8 + j (+16 for hi 8))
            const int ciB = kb + halfl * 8;
            const __hip_bfloat16* bp0 = &sW[nb + l16][ciB];
            b0.u[0] = *reinterpret_cast<const uint4*>(bp0);
            b0.u[1] = *reinterpret_cast<const uint4*>(bp0 + 16);
            const __hip_bfloat16* bp1 = &sW[nb + 16 + l16][ciB];
            b1.u[0] = *reinterpret_cast<const uint4*>(bp1);
            b1.u[1] = *reinterpret_cast<const uint4*>(bp1 + 16);
            // 4 independent WMMAs -> hazard slots covered by each other
            c00 = __builtin_amdgcn_wmma_f32_16x16x32_bf16(
                      false, a0.v, false, b0.v, (short)0, c00, false, false);
            c01 = __builtin_amdgcn_wmma_f32_16x16x32_bf16(
                      false, a0.v, false, b1.v, (short)0, c01, false, false);
            c10 = __builtin_amdgcn_wmma_f32_16x16x32_bf16(
                      false, a1.v, false, b0.v, (short)0, c10, false, false);
            c11 = __builtin_amdgcn_wmma_f32_16x16x32_bf16(
                      false, a1.v, false, b1.v, (short)0, c11, false, false);
        }
    }

    // ---- epilogue: y = relu(acc*alpha + beta), streamed (no reuse) ---------
    const float* betab = beta + b * F_;
    const int hrow = h0 + g;
    #pragma unroll
    for (int mi = 0; mi < 2; ++mi) {
        #pragma unroll
        for (int ni = 0; ni < 2; ++ni) {
            v8f cc = mi ? (ni ? c11 : c10) : (ni ? c01 : c00);
            const int n = nb + ni * 16 + l16;
            const float al = alpha[n];
            const float be = betab[n];
            #pragma unroll
            for (int r = 0; r < 8; ++r) {
                int wout = strip * 32 + mi * 16 + r + 8 * halfl;  // M = r+8*half
                float v = fmaxf(cc[r] * al + be, 0.f);
                __builtin_nontemporal_store(
                    v, &out[(((size_t)b * H_ + hrow) * W_ + wout) * F_ + n]);
            }
        }
    }
}

// ------------------------------------------------------------------ launch ---
extern "C" void kernel_launch(void* const* d_in, const int* in_sizes, int n_in,
                              void* d_out, int out_size, void* d_ws, size_t ws_size,
                              hipStream_t stream) {
    const float* x     = (const float*)d_in[0];
    const float* Wr    = (const float*)d_in[1];
    const float* Wa    = (const float*)d_in[2];
    const float* Wk    = (const float*)d_in[3];
    const float* bias  = (const float*)d_in[4];
    const float* scale = (const float*)d_in[5];
    const float* bnb   = (const float*)d_in[6];
    const float* mean  = (const float*)d_in[7];
    const float* var   = (const float*)d_in[8];
    float* out = (float*)d_out;

    char* ws = (char*)d_ws;
    float* attn  = (float*)(ws);                    // 32*4 f32
    float* alpha = (float*)(ws + 512);              // 128 f32
    float* beta  = (float*)(ws + 1024);             // 32*128 f32
    __hip_bfloat16* Wt = (__hip_bfloat16*)(ws + 32768);  // 32*9*128*128 bf16

    routing_kernel<<<B_, 256, 0, stream>>>(x, Wr, Wa, attn);
    mix_kernel<<<B_ * 9, 256, 0, stream>>>(Wk, attn, Wt);
    prep_kernel<<<B_, 128, 0, stream>>>(attn, bias, scale, bnb, mean, var, beta, alpha);
    condconv_wmma_kernel<<<dim3(2, H_ / 2, B_), 256, 0, stream>>>(x, Wt, alpha, beta, out);
}
